// VAE_71983651881070
// MI455X (gfx1250) — compile-verified, matrix operand order checked by reading, not run on hardware
//
#include <hip/hip_runtime.h>
#include <hip/hip_bf16.h>
#include <math.h>

// ---------------------------------------------------------------------------
// VAE forward (ELBO mean) for MI455X / gfx1250, wave32 + WMMA f16->f32.
// B=65536, D=512, H=1024, L=64.
//
// v2.2: bound K-loop unrolling (unroll_count(2)) + __launch_bounds__(256,2)
// to kill the scratch spills the full unroll caused (v2.1 allocated >500
// VGPRs and reloaded A-fragments from scratch inside the WMMA loop).
// ---------------------------------------------------------------------------

#define Bn 65536
#define Dn 512
#define Hn 1024
#define Ln 64
#define RT 64          // rows per block
#define MT 4           // 16-row M-tiles per block
#define NTHREADS 256
#define NWAVES 8
#define LOG2PI 1.8378770664093453f

#if __has_builtin(__builtin_amdgcn_global_load_async_to_lds_b128) && \
    __has_builtin(__builtin_amdgcn_s_wait_asynccnt)
#define ASYNC_LDS 1
#endif

typedef __attribute__((ext_vector_type(16))) _Float16 v16h;
typedef __attribute__((ext_vector_type(8)))  _Float16 v8h;
typedef __attribute__((ext_vector_type(8)))  float    v8f;
typedef __attribute__((ext_vector_type(4)))  int      v4i;

union Frag16 { v16h v; v8h h[2]; };

// WMMA D = A(16x32 f16) * B(32x16 f16) + C(16x16 f32)
__device__ __forceinline__ v8f wmma16(v16h a, v16h b, v8f c) {
  return __builtin_amdgcn_wmma_f32_16x16x32_f16(
      false, a, false, b, (short)0, c, false, false);
}

// One 16x32 fp16 fragment (A layout; B uses the same lane pattern with the
// weights stored transposed [N][K]).  ISA 7.12.2: lanes 0-15 hold
// K {k0+0..7, k0+16..23}; lanes 16-31 hold K {k0+8..15, k0+24..31}.
__device__ __forceinline__ v16h load_frag(const _Float16* base, int ldk, int k0) {
  const int lane = threadIdx.x & 31;
  const int r    = lane & 15;
  const int kb   = k0 + ((lane >> 4) << 3);
  Frag16 f;
  f.h[0] = *(const v8h*)(base + r * ldk + kb);
  f.h[1] = *(const v8h*)(base + r * ldk + kb + 16);
  return f.v;
}

// ---------------------------------------------------------------------------
// Weight prep: fp32 [R][C] row-major -> fp16 [C][R] (transposed, K contiguous)
// ---------------------------------------------------------------------------
__global__ void convert_transpose(const float* __restrict__ src,
                                  _Float16* __restrict__ dst, int R, int C) {
  int idx = blockIdx.x * blockDim.x + threadIdx.x;
  if (idx >= R * C) return;
  int r = idx / C, c = idx % C;
  dst[(size_t)c * R + r] = (_Float16)src[(size_t)r * C + c];
}

__global__ void zero_scalar(float* out) { out[0] = 0.0f; }

// ---------------------------------------------------------------------------
// Encoder: h = tanh(xW+b); mu,lv = hW+b; z = mu + exp(.5 lv)*eps; kl per row.
// ---------------------------------------------------------------------------
__global__ __launch_bounds__(NTHREADS, 2) void vae_encoder(
    const float* __restrict__ x, const float* __restrict__ eps,
    const _Float16* __restrict__ WtEncH, const float* __restrict__ bEncH,
    const _Float16* __restrict__ WtEncMu, const float* __restrict__ bEncMu,
    const _Float16* __restrict__ WtEncLv, const float* __restrict__ bEncLv,
    _Float16* __restrict__ zbuf, float* __restrict__ klbuf) {
  // pool: phase1 = xA (RT x Dn fp16, 64KB); phase2+ = muz|lvz (2 x 16KB)
  __shared__ __align__(16) char pool[RT * Dn * 2];
  __shared__ _Float16 hT[RT * Hn];                // 128 KB
  _Float16* xA  = (_Float16*)pool;
  float*    muz = (float*)pool;                   // RT*Ln f32
  float*    lvz = (float*)(pool + RT * Ln * 4);

  const int row0 = blockIdx.x * RT;
  const int tid  = threadIdx.x;
  const int wave = tid >> 5;
  const int lane = tid & 31;

  // Stage x tile -> fp16 LDS (coalesced float4 loads).
  {
    const float4* xs = (const float4*)(x + (size_t)row0 * Dn);
    for (int i = tid; i < (RT * Dn) / 4; i += NTHREADS) {
      float4 v = xs[i];
      int o = i * 4;
      xA[o + 0] = (_Float16)v.x; xA[o + 1] = (_Float16)v.y;
      xA[o + 2] = (_Float16)v.z; xA[o + 3] = (_Float16)v.w;
    }
  }
  __syncthreads();

  // ---- h = tanh(x @ W_enc_h + b) : N=1024, K=512; 1 B-frag -> 4 WMMA ----
  for (int nt = wave; nt < Hn / 16; nt += NWAVES) {
    const int n0 = nt * 16;
    const _Float16* wrow = WtEncH + (size_t)n0 * Dn;
    v8f acc[MT];
#pragma unroll
    for (int mt = 0; mt < MT; ++mt) acc[mt] = (v8f){};
#pragma clang loop unroll_count(2)
    for (int k0 = 0; k0 < Dn; k0 += 32) {
      __builtin_prefetch(wrow + k0 + 256, 0, 3);  // keep in near caches
      v16h b = load_frag(wrow, Dn, k0);
#pragma unroll
      for (int mt = 0; mt < MT; ++mt) {
        v16h a = load_frag(xA + mt * 16 * Dn, Dn, k0);
        acc[mt] = wmma16(a, b, acc[mt]);
      }
    }
    const int n  = n0 + (lane & 15);
    const int mb = (lane >> 4) * 8;
    const float bias = bEncH[n];
#pragma unroll
    for (int mt = 0; mt < MT; ++mt)
#pragma unroll
      for (int i = 0; i < 8; ++i)
        hT[(mt * 16 + mb + i) * Hn + n] = (_Float16)tanhf(acc[mt][i] + bias);
  }
  __syncthreads();

  // ---- mu_z / lv_z : 8 jobs (4 mu-tiles + 4 lv-tiles), one per wave ----
  {
    const int j = wave;
    const bool isMu = (j < 4);
    const int n0 = (j & 3) * 16;
    const _Float16* Wt  = isMu ? WtEncMu : WtEncLv;
    const float*    bia = isMu ? bEncMu  : bEncLv;
    float*          oT  = isMu ? muz     : lvz;
    const _Float16* wrow = Wt + (size_t)n0 * Hn;
    v8f acc[MT];
#pragma unroll
    for (int mt = 0; mt < MT; ++mt) acc[mt] = (v8f){};
#pragma clang loop unroll_count(2)
    for (int k0 = 0; k0 < Hn; k0 += 32) {
      v16h b = load_frag(wrow, Hn, k0);
#pragma unroll
      for (int mt = 0; mt < MT; ++mt) {
        v16h a = load_frag(hT + mt * 16 * Hn, Hn, k0);
        acc[mt] = wmma16(a, b, acc[mt]);
      }
    }
    const int n  = n0 + (lane & 15);
    const int mb = (lane >> 4) * 8;
    const float bv = bia[n];
#pragma unroll
    for (int mt = 0; mt < MT; ++mt)
#pragma unroll
      for (int i = 0; i < 8; ++i)
        oT[(mt * 16 + mb + i) * Ln + n] = acc[mt][i] + bv;
  }
  __syncthreads();

  // ---- z = mu + exp(0.5 lv)*eps ; kl = -0.5 sum(1+lv-mu^2-exp(lv)) ----
  if (tid < RT) {
    const int r = tid, grow = row0 + r;
    float kl = 0.0f;
    for (int l = 0; l < Ln; ++l) {
      float mu = muz[r * Ln + l];
      float lv = lvz[r * Ln + l];
      float ev = eps[(size_t)grow * Ln + l];
      zbuf[(size_t)grow * Ln + l] = (_Float16)(mu + __expf(0.5f * lv) * ev);
      kl += 1.0f + lv - mu * mu - __expf(lv);
    }
    klbuf[grow] = -0.5f * kl;
  }
}

// ---------------------------------------------------------------------------
// Decoder: hd = tanh(zW+b); softmax(hd W_mu + b); lv_x GEMM fused into the
// diagonal-Gaussian NLL; scalar ELBO via global f32 atomics.
// ---------------------------------------------------------------------------
__global__ __launch_bounds__(NTHREADS, 2) void vae_decoder(
    const float* __restrict__ x,
    const _Float16* __restrict__ zbuf,
    const _Float16* __restrict__ WtDecH, const float* __restrict__ bDecH,
    const _Float16* __restrict__ WtDecMu, const float* __restrict__ bDecMu,
    const _Float16* __restrict__ WtDecLv, const float* __restrict__ bDecLv,
    const float* __restrict__ klbuf, float* __restrict__ out) {
  __shared__ __align__(16) _Float16 zT[RT * Ln];  //   8 KB
  __shared__ _Float16 hdT[RT * Hn];               // 128 KB
  __shared__ _Float16 pT[RT * Dn];                //  64 KB: logits -> exp
  __shared__ float red[4][RT];
  __shared__ float rowmax[RT];
  __shared__ float invsum[RT];
  __shared__ float rowNLL[RT];

  const int row0 = blockIdx.x * RT;
  const int tid  = threadIdx.x;
  const int wave = tid >> 5;
  const int lane = tid & 31;

  if (tid < RT) rowNLL[tid] = 0.0f;

  // ---- stage z tile (RT x Ln fp16 = 8KB) into LDS, async if available ----
  {
    const char* zsrc = (const char*)(zbuf + (size_t)row0 * Ln);
#ifdef ASYNC_LDS
    typedef __attribute__((address_space(1))) v4i* gp_t;   // global int4*
    typedef __attribute__((address_space(3))) v4i* lp_t;   // LDS int4*
    for (int i = tid; i < (RT * Ln * 2) / 16; i += NTHREADS) {
      gp_t g = (gp_t)(unsigned long long)(zsrc + i * 16);
      lp_t l = (lp_t)(unsigned)(unsigned long long)((char*)zT + i * 16);
      __builtin_amdgcn_global_load_async_to_lds_b128(g, l, 0, 0);
    }
    __builtin_amdgcn_s_wait_asynccnt(0);
#else
    for (int i = tid; i < (RT * Ln) / 8; i += NTHREADS)
      ((v8h*)zT)[i] = ((const v8h*)zsrc)[i];
#endif
  }
  __syncthreads();

  // ---- hd = tanh(z @ W_dec_h + b) : N=1024, K=64 ----
  for (int nt = wave; nt < Hn / 16; nt += NWAVES) {
    const int n0 = nt * 16;
    const _Float16* wrow = WtDecH + (size_t)n0 * Ln;
    v8f acc[MT];
#pragma unroll
    for (int mt = 0; mt < MT; ++mt) acc[mt] = (v8f){};
#pragma clang loop unroll_count(1)
    for (int k0 = 0; k0 < Ln; k0 += 32) {
      v16h b = load_frag(wrow, Ln, k0);
#pragma unroll
      for (int mt = 0; mt < MT; ++mt) {
        v16h a = load_frag(zT + mt * 16 * Ln, Ln, k0);
        acc[mt] = wmma16(a, b, acc[mt]);
      }
    }
    const int n  = n0 + (lane & 15);
    const int mb = (lane >> 4) * 8;
    const float bias = bDecH[n];
#pragma unroll
    for (int mt = 0; mt < MT; ++mt)
#pragma unroll
      for (int i = 0; i < 8; ++i)
        hdT[(mt * 16 + mb + i) * Hn + n] = (_Float16)tanhf(acc[mt][i] + bias);
  }
  __syncthreads();

  // ---- logits = hd @ W_dec_mu + b : N=512, K=1024 ----
  for (int nt = wave; nt < Dn / 16; nt += NWAVES) {
    const int n0 = nt * 16;
    const _Float16* wrow = WtDecMu + (size_t)n0 * Hn;
    v8f acc[MT];
#pragma unroll
    for (int mt = 0; mt < MT; ++mt) acc[mt] = (v8f){};
#pragma clang loop unroll_count(2)
    for (int k0 = 0; k0 < Hn; k0 += 32) {
      __builtin_prefetch(wrow + k0 + 512, 0, 3);
      v16h b = load_frag(wrow, Hn, k0);
#pragma unroll
      for (int mt = 0; mt < MT; ++mt) {
        v16h a = load_frag(hdT + mt * 16 * Hn, Hn, k0);
        acc[mt] = wmma16(a, b, acc[mt]);
      }
    }
    const int n  = n0 + (lane & 15);
    const int mb = (lane >> 4) * 8;
    const float bias = bDecMu[n];
#pragma unroll
    for (int mt = 0; mt < MT; ++mt)
#pragma unroll
      for (int i = 0; i < 8; ++i)
        pT[(mt * 16 + mb + i) * Dn + n] = (_Float16)(acc[mt][i] + bias);
  }
  __syncthreads();

  // ---- row softmax over 512 logits: 4 threads/row x 128 elements ----
  {
    const int r = tid & (RT - 1), c = tid >> 6;  // c in 0..3
    float m = -1e30f;
    for (int j = 0; j < 128; ++j)
      m = fmaxf(m, (float)pT[r * Dn + c * 128 + j]);
    red[c][r] = m;
    __syncthreads();
    if (tid < RT) {
      float mm = fmaxf(fmaxf(red[0][tid], red[1][tid]),
                       fmaxf(red[2][tid], red[3][tid]));
      rowmax[tid] = mm;
    }
    __syncthreads();
    const float rm = rowmax[r];
    float s = 0.0f;
    for (int j = 0; j < 128; ++j) {
      float e = __expf((float)pT[r * Dn + c * 128 + j] - rm);
      pT[r * Dn + c * 128 + j] = (_Float16)e;  // store numerator
      s += e;
    }
    __syncthreads();
    red[c][r] = s;
    __syncthreads();
    if (tid < RT)
      invsum[tid] = 1.0f / (red[0][tid] + red[1][tid] + red[2][tid] + red[3][tid]);
    __syncthreads();
  }

  // ---- lv_x = hd @ W_dec_lv + b fused into NLL accumulation ----
  {
    float part[MT][8];
#pragma unroll
    for (int mt = 0; mt < MT; ++mt)
#pragma unroll
      for (int i = 0; i < 8; ++i) part[mt][i] = 0.0f;

    for (int nt = wave; nt < Dn / 16; nt += NWAVES) {
      const int n0 = nt * 16;
      const _Float16* wrow = WtDecLv + (size_t)n0 * Hn;
      v8f acc[MT];
#pragma unroll
      for (int mt = 0; mt < MT; ++mt) acc[mt] = (v8f){};
#pragma clang loop unroll_count(2)
      for (int k0 = 0; k0 < Hn; k0 += 32) {
        __builtin_prefetch(wrow + k0 + 512, 0, 3);
        v16h b = load_frag(wrow, Hn, k0);
#pragma unroll
        for (int mt = 0; mt < MT; ++mt) {
          v16h a = load_frag(hdT + mt * 16 * Hn, Hn, k0);
          acc[mt] = wmma16(a, b, acc[mt]);
        }
      }
      const int n  = n0 + (lane & 15);
      const int mb = (lane >> 4) * 8;
      const float bias = bDecLv[n];
#pragma unroll
      for (int mt = 0; mt < MT; ++mt)
#pragma unroll
        for (int i = 0; i < 8; ++i) {
          const int m = mt * 16 + mb + i;
          float lv  = acc[mt][i] + bias;
          float mux = (float)pT[m * Dn + n] * invsum[m];
          float xv  = x[(size_t)(row0 + m) * Dn + n];
          float d   = xv - mux;
          part[mt][i] += lv + d * d * __expf(-lv) + LOG2PI;
        }
    }
    const int mb = (lane >> 4) * 8;
#pragma unroll
    for (int mt = 0; mt < MT; ++mt)
#pragma unroll
      for (int i = 0; i < 8; ++i)
        atomicAdd(&rowNLL[mt * 16 + mb + i], part[mt][i]);
  }
  __syncthreads();

  if (tid < RT) {
    const int grow = row0 + tid;
    float lnp = -0.5f * rowNLL[tid];
    atomicAdd(out, (lnp - klbuf[grow]) * (1.0f / (float)Bn));
  }
}

// ---------------------------------------------------------------------------
// Host-side launch
// ---------------------------------------------------------------------------
extern "C" void kernel_launch(void* const* d_in, const int* in_sizes, int n_in,
                              void* d_out, int out_size, void* d_ws, size_t ws_size,
                              hipStream_t stream) {
  const float* x        = (const float*)d_in[0];
  const float* eps      = (const float*)d_in[1];
  const float* W_enc_h  = (const float*)d_in[2];
  const float* b_enc_h  = (const float*)d_in[3];
  const float* W_enc_mu = (const float*)d_in[4];
  const float* b_enc_mu = (const float*)d_in[5];
  const float* W_enc_lv = (const float*)d_in[6];
  const float* b_enc_lv = (const float*)d_in[7];
  const float* W_dec_h  = (const float*)d_in[8];
  const float* b_dec_h  = (const float*)d_in[9];
  const float* W_dec_mu = (const float*)d_in[10];
  const float* b_dec_mu = (const float*)d_in[11];
  const float* W_dec_lv = (const float*)d_in[12];
  const float* b_dec_lv = (const float*)d_in[13];

  // Workspace carve-up (all offsets 256B-aligned).
  char* ws = (char*)d_ws;
  _Float16* WtEncH  = (_Float16*)ws;  ws += (size_t)Hn * Dn * 2;  // [1024][512]
  _Float16* WtEncMu = (_Float16*)ws;  ws += (size_t)Ln * Hn * 2;  // [64][1024]
  _Float16* WtEncLv = (_Float16*)ws;  ws += (size_t)Ln * Hn * 2;
  _Float16* WtDecH  = (_Float16*)ws;  ws += (size_t)Hn * Ln * 2;  // [1024][64]
  _Float16* WtDecMu = (_Float16*)ws;  ws += (size_t)Dn * Hn * 2;  // [512][1024]
  _Float16* WtDecLv = (_Float16*)ws;  ws += (size_t)Dn * Hn * 2;
  _Float16* zbuf    = (_Float16*)ws;  ws += (size_t)Bn * Ln * 2;  // [B][64]
  float*    klbuf   = (float*)ws;     ws += (size_t)Bn * 4;       // [B]

  auto cvt = [&](const float* s, _Float16* d, int R, int C) {
    int total = R * C;
    convert_transpose<<<(total + 255) / 256, 256, 0, stream>>>(s, d, R, C);
  };
  cvt(W_enc_h,  WtEncH,  Dn, Hn);
  cvt(W_enc_mu, WtEncMu, Hn, Ln);
  cvt(W_enc_lv, WtEncLv, Hn, Ln);
  cvt(W_dec_h,  WtDecH,  Ln, Hn);
  cvt(W_dec_mu, WtDecMu, Hn, Dn);
  cvt(W_dec_lv, WtDecLv, Hn, Dn);

  zero_scalar<<<1, 1, 0, stream>>>((float*)d_out);

  vae_encoder<<<Bn / RT, NTHREADS, 0, stream>>>(
      x, eps, WtEncH, b_enc_h, WtEncMu, b_enc_mu, WtEncLv, b_enc_lv,
      zbuf, klbuf);

  vae_decoder<<<Bn / RT, NTHREADS, 0, stream>>>(
      x, zbuf, WtDecH, b_dec_h, WtDecMu, b_dec_mu, WtDecLv, b_dec_lv,
      klbuf, (float*)d_out);
}
